// InfiniAttention_14602888806987
// MI455X (gfx1250) — matrix operand order repaired and checked by reading, not compiled, not run on hardware
//
#include <hip/hip_runtime.h>

// ---------------------------------------------------------------------------
// InfiniAttention for MI455X (gfx1250, wave32, WMMA bf16 16x16x32)
//   B=4, S=4096, E=1024, H=16, D=64, NSEG=16, L=256
// Pipeline:
//   1) convert x, Wq..Wo to bf16 (workspace)
//   2) bf16 WMMA GEMM (async global->LDS staging, double buffered)
//   3) attention scan: 1 workgroup per (b,h), LDS-resident memory/z state
//   4) bf16 WMMA GEMM: out = attn @ Wo^T + bo -> f32
// ---------------------------------------------------------------------------

#define B_ 4
#define S_ 4096
#define E_ 1024
#define H_ 16
#define D_ 64
#define NSEG_ 16
#define L_ 256

typedef __attribute__((ext_vector_type(16))) __bf16 bf16x16;
typedef __attribute__((ext_vector_type(8)))  float  floatx8;

union Frag { unsigned u[8]; bf16x16 v; };

__device__ __forceinline__ unsigned short f2bf(float f) {
  union { float f; unsigned u; } v; v.f = f;
  unsigned r = v.u + 0x7FFFu + ((v.u >> 16) & 1u);   // round-to-nearest-even
  return (unsigned short)(r >> 16);
}
__device__ __forceinline__ float bf2f(unsigned short h) {
  union { unsigned u; float f; } v; v.u = ((unsigned)h) << 16;
  return v.f;
}
__device__ __forceinline__ unsigned pack2(float lo, float hi) {
  return (unsigned)f2bf(lo) | ((unsigned)f2bf(hi) << 16);
}
__device__ __forceinline__ floatx8 zero8() {
  floatx8 v;
#pragma unroll
  for (int i = 0; i < 8; ++i) v[i] = 0.f;
  return v;
}
__device__ __forceinline__ floatx8 wmma_bf16(const Frag& a, const Frag& b, floatx8 c) {
  return __builtin_amdgcn_wmma_f32_16x16x32_bf16(false, a.v, false, b.v,
                                                 (short)0, c, false, false);
}
// A-fragment (16x32 bf16): lane<16 row=lane, 16 contiguous K starting at 2*hk
__device__ __forceinline__ void load_a_frag(Frag& f, const unsigned short* rowp, int hk) {
#pragma unroll
  for (int i = 0; i < 4; ++i) f.u[i]     = *(const unsigned*)(rowp + 2 * hk + 2 * i);
#pragma unroll
  for (int i = 0; i < 4; ++i) f.u[4 + i] = *(const unsigned*)(rowp + 2 * hk + 8 + 2 * i);
}
// B-fragment (32x16 bf16): lane holds 16 contiguous K for its N column
__device__ __forceinline__ void load_b_frag(Frag& f, const unsigned short* p) {
#pragma unroll
  for (int i = 0; i < 8; ++i) f.u[i] = *(const unsigned*)(p + 2 * i);
}

// ---------------------------------------------------------------------------
// CDNA5 async global->LDS copy (ASYNCcnt-tracked), with sync fallback
//   probe-learned signature: (int4* AS1 src, int4* AS3 dst, imm offset, imm cpol)
// ---------------------------------------------------------------------------
#if defined(__gfx1250__) && __has_builtin(__builtin_amdgcn_global_load_async_to_lds_b128)
#define HAVE_ASYNC_LDS 1
#else
#define HAVE_ASYNC_LDS 0
#endif

#if HAVE_ASYNC_LDS
typedef int v4i_ __attribute__((vector_size(16)));
typedef __attribute__((address_space(1))) v4i_* as1_v4i;
typedef __attribute__((address_space(3))) v4i_* as3_v4i;
#endif

__device__ __forceinline__ void async_b128(const unsigned short* g, unsigned short* l) {
#if HAVE_ASYNC_LDS
  __builtin_amdgcn_global_load_async_to_lds_b128((as1_v4i)(g), (as3_v4i)(l), 0, 0);
#else
  *(uint4*)l = *(const uint4*)g;
#endif
}
__device__ __forceinline__ void wait_async() {
#if HAVE_ASYNC_LDS
#if __has_builtin(__builtin_amdgcn_s_wait_asynccnt)
  __builtin_amdgcn_s_wait_asynccnt(0);
#else
  asm volatile("s_wait_asynccnt 0" ::: "memory");
#endif
#endif
}

// ---------------------------------------------------------------------------
// fp32 -> bf16 conversion, 4 elements/thread
// ---------------------------------------------------------------------------
__global__ __launch_bounds__(256) void cvt_f32_bf16(const float* __restrict__ in,
                                                    unsigned short* __restrict__ out,
                                                    int n) {
  int i = (blockIdx.x * 256 + threadIdx.x) * 4;
  if (i + 3 < n) {
    float4 v = *(const float4*)(in + i);
    unsigned w0 = pack2(v.x, v.y), w1 = pack2(v.z, v.w);
    *(unsigned*)(out + i)     = w0;
    *(unsigned*)(out + i + 2) = w1;
  }
}

// ---------------------------------------------------------------------------
// Generic NT GEMM: C[m,n] = sum_k A[m,k]*Bw[n,k] + bias[n]
//   A: MxK bf16 row-major, Bw: NxK bf16 row-major. Block tile 128x128,
//   K-step 32, double-buffered async global->LDS staging.
//   8 waves as 2(M)x4(N); each wave: 64x32 -> 4x2 WMMA tiles.
// ---------------------------------------------------------------------------
template <bool OUT_BF16>
__global__ __launch_bounds__(256) void gemm_nt(const unsigned short* __restrict__ A,
                                               const unsigned short* __restrict__ Bw,
                                               const float* __restrict__ bias,
                                               void* __restrict__ Cout,
                                               int M, int N, int K) {
  __shared__ unsigned short As[2][128][40];
  __shared__ unsigned short Bs[2][128][40];

  const int tid  = threadIdx.x;
  const int lane = tid & 31;
  const int wave = tid >> 5;
  const int bm = blockIdx.x * 128;
  const int bn = blockIdx.y * 128;
  const int wm = (wave >> 2) * 64;
  const int wn = (wave & 3) * 32;

  floatx8 acc[4][2];
#pragma unroll
  for (int i = 0; i < 4; ++i)
#pragma unroll
    for (int j = 0; j < 2; ++j) acc[i][j] = zero8();

  const int srow = tid >> 1;             // 0..127
  const int scol = (tid & 1) * 16;       // 0 or 16
  const int lr = lane & 15;
  const int hk = (lane >> 4) * 8;
  const int kb = (lane >> 4) * 16;

  // async-stage one 128x32 K-tile of A and B into LDS buffer `buf`
  auto stage = [&](int kt, int buf) {
    const unsigned short* ga = A  + (size_t)(bm + srow) * K + kt * 32 + scol;
    const unsigned short* gb = Bw + (size_t)(bn + srow) * K + kt * 32 + scol;
    async_b128(ga,     &As[buf][srow][scol]);
    async_b128(ga + 8, &As[buf][srow][scol + 8]);
    async_b128(gb,     &Bs[buf][srow][scol]);
    async_b128(gb + 8, &Bs[buf][srow][scol + 8]);
  };

  const int KT = K >> 5;
  stage(0, 0);
  wait_async();
  __syncthreads();

  for (int kt = 0; kt < KT; ++kt) {
    const int cur = kt & 1;
    if (kt + 1 < KT) stage(kt + 1, cur ^ 1);   // prefetch next K-tile

    Frag fb[2];
#pragma unroll
    for (int nsub = 0; nsub < 2; ++nsub)
      load_b_frag(fb[nsub], &Bs[cur][wn + nsub * 16 + lr][kb]);
#pragma unroll
    for (int msub = 0; msub < 4; ++msub) {
      Frag fa;
      load_a_frag(fa, &As[cur][wm + msub * 16 + lr][0], hk);
#pragma unroll
      for (int nsub = 0; nsub < 2; ++nsub)
        acc[msub][nsub] = wmma_bf16(fa, fb[nsub], acc[msub][nsub]);
    }

    wait_async();        // prefetched tile has landed
    __syncthreads();     // everyone done reading `cur`, next tile published
  }

  // epilogue: D-layout rows = msub*16 + r + 8*(lane>=16), col = lane&15
#pragma unroll
  for (int msub = 0; msub < 4; ++msub) {
#pragma unroll
    for (int nsub = 0; nsub < 2; ++nsub) {
      int col = bn + wn + nsub * 16 + (lane & 15);
      float bv = bias[col];
#pragma unroll
      for (int r = 0; r < 8; ++r) {
        int row = bm + wm + msub * 16 + r + ((lane >> 4) * 8);
        float v = acc[msub][nsub][r] + bv;
        if (OUT_BF16)
          ((unsigned short*)Cout)[(size_t)row * N + col] = f2bf(v);
        else
          ((float*)Cout)[(size_t)row * N + col] = v;
      }
    }
  }
}

// ---------------------------------------------------------------------------
// Attention scan: one block per (b,h); 8 waves; loop over 16 segments.
// LDS: qs/sq/ks tiles (256x72 bf16), skT/vsT (64x264 bf16), memT (64x68 f32),
//      zrow(64), sqsum(256), per-wave P scratch (16x40 bf16).
// memT[e][d] == memory[d][e]  (so all WMMA B-operands read row-contiguous)
// ---------------------------------------------------------------------------
__global__ __launch_bounds__(256) void attn_scan(const unsigned short* __restrict__ qb,
                                                 const unsigned short* __restrict__ kb,
                                                 const unsigned short* __restrict__ vb,
                                                 const float* __restrict__ beta,
                                                 unsigned short* __restrict__ attnb) {
  extern __shared__ char smem_raw[];
  unsigned short* qsm  = (unsigned short*)smem_raw;   // [256][72]
  unsigned short* sqm  = qsm + 256 * 72;              // [256][72]
  unsigned short* ksm  = sqm + 256 * 72;              // [256][72]
  unsigned short* skTm = ksm + 256 * 72;              // [64][264]
  unsigned short* vsTm = skTm + 64 * 264;             // [64][264]
  float* memT  = (float*)(vsTm + 64 * 264);           // [64][68]
  float* zrow  = memT + 64 * 68;                      // [64]
  float* sqsum = zrow + 64;                           // [256]
  unsigned short* Pscr = (unsigned short*)(sqsum + 256); // [8][16][40]

  const int bh = blockIdx.x;
  const int b = bh >> 4, h = bh & 15;
  const size_t segbase0 = (size_t)b * S_ * E_ + (size_t)h * (NSEG_ * L_ * D_);
  const int tid  = threadIdx.x;
  const int lane = tid & 31;
  const int wave = tid >> 5;
  const int lr = lane & 15;
  const int hk = (lane >> 4) * 8;
  const int kb16 = (lane >> 4) * 16;
  const float gate = 1.f / (1.f + __expf(-beta[0]));

  for (int i = tid; i < 64 * 68; i += 256) memT[i] = 0.f;
  if (tid < 64) zrow[tid] = 0.f;
  __syncthreads();

  for (int seg = 0; seg < NSEG_; ++seg) {
    const size_t base = segbase0 + (size_t)seg * (L_ * D_);

    // ---------------- phase 1: load + elu preprocess (one row / thread) ----
    {
      const int l = tid;
      const uint4* qp = (const uint4*)(qb + base + (size_t)l * D_);
      const uint4* kp = (const uint4*)(kb + base + (size_t)l * D_);
      const uint4* vp = (const uint4*)(vb + base + (size_t)l * D_);
      float ssum = 0.f;
#pragma unroll
      for (int c = 0; c < 8; ++c) {
        uint4 a = qp[c];
        unsigned w[4] = {a.x, a.y, a.z, a.w};
#pragma unroll
        for (int j = 0; j < 4; ++j) {
          *(unsigned*)&qsm[l * 72 + c * 8 + j * 2] = w[j];   // raw q
          float v0 = bf2f((unsigned short)(w[j] & 0xffff));
          float v1 = bf2f((unsigned short)(w[j] >> 16));
          float e0 = (v0 > 0.f ? v0 : __expf(v0) - 1.f) + 1.f;
          float e1 = (v1 > 0.f ? v1 : __expf(v1) - 1.f) + 1.f;
          ssum += e0 + e1;
          *(unsigned*)&sqm[l * 72 + c * 8 + j * 2] = pack2(e0, e1);
        }
      }
      sqsum[l] = ssum;
#pragma unroll
      for (int c = 0; c < 8; ++c) {
        uint4 a = kp[c];
        unsigned w[4] = {a.x, a.y, a.z, a.w};
#pragma unroll
        for (int j = 0; j < 4; ++j) {
          *(unsigned*)&ksm[l * 72 + c * 8 + j * 2] = w[j];   // raw k
          int d = c * 8 + j * 2;
          float v0 = bf2f((unsigned short)(w[j] & 0xffff));
          float v1 = bf2f((unsigned short)(w[j] >> 16));
          float e0 = (v0 > 0.f ? v0 : __expf(v0) - 1.f) + 1.f;
          float e1 = (v1 > 0.f ? v1 : __expf(v1) - 1.f) + 1.f;
          skTm[(d)     * 264 + l] = f2bf(e0);                // sk^T
          skTm[(d + 1) * 264 + l] = f2bf(e1);
        }
      }
#pragma unroll
      for (int c = 0; c < 8; ++c) {
        uint4 a = vp[c];
        unsigned w[4] = {a.x, a.y, a.z, a.w};
#pragma unroll
        for (int j = 0; j < 4; ++j) {
          int d = c * 8 + j * 2;
          vsTm[(d)     * 264 + l] = (unsigned short)(w[j] & 0xffff); // v^T
          vsTm[(d + 1) * 264 + l] = (unsigned short)(w[j] >> 16);
        }
      }
    }
    __syncthreads();

    // ---------------- phase 2: A_mem + causal-softmax A_dot + combine ------
    {
      int mt[2] = {wave, 15 - wave};
#pragma unroll 1
      for (int mi = 0; mi < 2; ++mi) {
        const int m = mt[mi];

        // ---- A_mem = sq @ memory  (B-op rows = memT[e][d] row-contig) ----
        floatx8 amem[4];
#pragma unroll
        for (int nt = 0; nt < 4; ++nt) amem[nt] = zero8();
#pragma unroll
        for (int kk = 0; kk < 2; ++kk) {
          Frag fa;
          load_a_frag(fa, &sqm[(m * 16 + lr) * 72 + kk * 32], hk);
#pragma unroll
          for (int nt = 0; nt < 4; ++nt) {
            Frag fb;
            const float* mr = &memT[(nt * 16 + lr) * 68 + kk * 32 + kb16];
#pragma unroll
            for (int i = 0; i < 8; ++i) fb.u[i] = pack2(mr[2 * i], mr[2 * i + 1]);
            amem[nt] = wmma_bf16(fa, fb, amem[nt]);
          }
        }

        // ---- A_dot: flash softmax over 32-key pairs ----
        floatx8 adot[4];
#pragma unroll
        for (int nt = 0; nt < 4; ++nt) adot[nt] = zero8();
        float rmax[8], rsum[8];
#pragma unroll
        for (int r = 0; r < 8; ++r) { rmax[r] = -1e30f; rsum[r] = 0.f; }

        const int npair = (m >> 1) + 1;
        const int row_off = m * 16 + ((lane >> 4) * 8);
        for (int jp = 0; jp < npair; ++jp) {
          floatx8 sc[2] = {zero8(), zero8()};
#pragma unroll
          for (int kk = 0; kk < 2; ++kk) {
            Frag fa;
            load_a_frag(fa, &qsm[(m * 16 + lr) * 72 + kk * 32], hk);
#pragma unroll
            for (int half = 0; half < 2; ++half) {
              Frag fb;
              load_b_frag(fb, &ksm[((2 * jp + half) * 16 + lr) * 72 + kk * 32 + kb16]);
              sc[half] = wmma_bf16(fa, fb, sc[half]);
            }
          }
          // mask + scale + online softmax (rows live across 16-lane halves)
#pragma unroll
          for (int r = 0; r < 8; ++r) {
            int row = row_off + r;
#pragma unroll
            for (int half = 0; half < 2; ++half) {
              int col = jp * 32 + half * 16 + (lane & 15);
              float s = sc[half][r] * 0.125f;               // 1/sqrt(D)
              if (col > row) s = -1e30f;                    // causal mask
              sc[half][r] = s;
            }
            float tm = fmaxf(sc[0][r], sc[1][r]);
#pragma unroll
            for (int off = 8; off >= 1; off >>= 1) tm = fmaxf(tm, __shfl_xor(tm, off));
            float nmax = fmaxf(rmax[r], tm);
            float rescale = __expf(rmax[r] - nmax);
            rmax[r] = nmax;
            float psum = 0.f;
#pragma unroll
            for (int half = 0; half < 2; ++half) {
              float p = __expf(sc[half][r] - nmax);
              sc[half][r] = p;
              psum += p;
            }
#pragma unroll
            for (int off = 8; off >= 1; off >>= 1) psum += __shfl_xor(psum, off);
            rsum[r] = rsum[r] * rescale + psum;
#pragma unroll
            for (int nt = 0; nt < 4; ++nt) adot[nt][r] *= rescale;
            // P -> wave-private scratch (D-layout -> A-layout bounce)
            int prow = r + ((lane >> 4) * 8);
#pragma unroll
            for (int half = 0; half < 2; ++half)
              Pscr[wave * 640 + prow * 40 + half * 16 + (lane & 15)] = f2bf(sc[half][r]);
          }
          // P @ V : A = P(16x32), B = vsT rows (d, key-run)
          Frag fp;
          load_a_frag(fp, &Pscr[wave * 640 + lr * 40], hk);
#pragma unroll
          for (int nt = 0; nt < 4; ++nt) {
            Frag fv;
            load_b_frag(fv, &vsTm[(nt * 16 + lr) * 264 + jp * 32 + kb16]);
            adot[nt] = wmma_bf16(fp, fv, adot[nt]);
          }
        }

        // ---- combine + store ----
#pragma unroll
        for (int nt = 0; nt < 4; ++nt) {
          int col = nt * 16 + (lane & 15);
          float zc = zrow[col];
#pragma unroll
          for (int r = 0; r < 8; ++r) {
            int row = row_off + r;
            float ad = adot[nt][r] / rsum[r];
            float den = sqsum[row] * zc + 1e-6f;
            float am = amem[nt][r] / den;
            float o = gate * am + (1.f - gate) * ad;
            attnb[base + (size_t)row * D_ + col] = f2bf(o);
          }
        }
      }
    }
    __syncthreads();

    // ---------------- phase 3: memT += vs^T @ sk ; zrow += colsum(sk) ------
    {
#pragma unroll 1
      for (int t = 0; t < 2; ++t) {
        int id = wave * 2 + t;
        int ti = id >> 2, tj = id & 3;
        floatx8 acc = zero8();
#pragma unroll
        for (int kt = 0; kt < 8; ++kt) {
          Frag fa;
          load_a_frag(fa, &vsTm[(ti * 16 + lr) * 264 + kt * 32], hk);
          Frag fb;
          load_b_frag(fb, &skTm[(tj * 16 + lr) * 264 + kt * 32 + kb16]);
          acc = wmma_bf16(fa, fb, acc);
        }
#pragma unroll
        for (int r = 0; r < 8; ++r) {
          int e = ti * 16 + r + ((lane >> 4) * 8);
          int d = tj * 16 + (lane & 15);
          memT[e * 68 + d] += acc[r];
        }
      }
      if (tid < 64) {
        float a = 0.f;
        const unsigned short* pr = &skTm[tid * 264];
#pragma unroll 4
        for (int i = 0; i < 128; ++i) {
          unsigned w = *(const unsigned*)(pr + 2 * i);
          a += bf2f((unsigned short)(w & 0xffff)) + bf2f((unsigned short)(w >> 16));
        }
        zrow[tid] += a;
      }
    }
    __syncthreads();
  }
}

// ---------------------------------------------------------------------------
extern "C" void kernel_launch(void* const* d_in, const int* in_sizes, int n_in,
                              void* d_out, int out_size, void* d_ws, size_t ws_size,
                              hipStream_t stream) {
  (void)in_sizes; (void)n_in; (void)out_size; (void)ws_size;
  const float* x    = (const float*)d_in[0];
  const float* Wq   = (const float*)d_in[1];
  const float* Wk   = (const float*)d_in[2];
  const float* Wv   = (const float*)d_in[3];
  const float* Wo   = (const float*)d_in[4];
  const float* bq   = (const float*)d_in[5];
  const float* bk   = (const float*)d_in[6];
  const float* bv   = (const float*)d_in[7];
  const float* bo   = (const float*)d_in[8];
  const float* beta = (const float*)d_in[9];

  const size_t NX = (size_t)B_ * S_ * E_;   // 16,777,216
  const size_t NW = (size_t)E_ * E_;        // 1,048,576

  char* w = (char*)d_ws;
  unsigned short* xb  = (unsigned short*)w; w += NX * 2;
  unsigned short* Wqb = (unsigned short*)w; w += NW * 2;
  unsigned short* Wkb = (unsigned short*)w; w += NW * 2;
  unsigned short* Wvb = (unsigned short*)w; w += NW * 2;
  unsigned short* Wob = (unsigned short*)w; w += NW * 2;
  unsigned short* qb  = (unsigned short*)w; w += NX * 2;
  unsigned short* kbf = (unsigned short*)w; w += NX * 2;
  unsigned short* vbf = (unsigned short*)w; w += NX * 2;
  unsigned short* att = (unsigned short*)w; w += NX * 2;

  // 1) fp32 -> bf16
  cvt_f32_bf16<<<dim3((unsigned)(NX / 1024)), 256, 0, stream>>>(x,  xb,  (int)NX);
  cvt_f32_bf16<<<dim3((unsigned)(NW / 1024)), 256, 0, stream>>>(Wq, Wqb, (int)NW);
  cvt_f32_bf16<<<dim3((unsigned)(NW / 1024)), 256, 0, stream>>>(Wk, Wkb, (int)NW);
  cvt_f32_bf16<<<dim3((unsigned)(NW / 1024)), 256, 0, stream>>>(Wv, Wvb, (int)NW);
  cvt_f32_bf16<<<dim3((unsigned)(NW / 1024)), 256, 0, stream>>>(Wo, Wob, (int)NW);

  // 2) QKV projections (M=16384, N=1024, K=1024), bf16 out
  const int M = B_ * S_;
  dim3 ggrid(M / 128, E_ / 128);
  gemm_nt<true><<<ggrid, 256, 0, stream>>>(xb, Wqb, bq, qb,  M, E_, E_);
  gemm_nt<true><<<ggrid, 256, 0, stream>>>(xb, Wkb, bk, kbf, M, E_, E_);
  gemm_nt<true><<<ggrid, 256, 0, stream>>>(xb, Wvb, bv, vbf, M, E_, E_);

  // 3) attention scan: one workgroup per (b,h); ~207 KB dynamic LDS
  const size_t smem = (size_t)(3 * 256 * 72 + 2 * 64 * 264) * 2   // bf16 tiles
                    + (size_t)(64 * 68 + 64 + 256) * 4            // f32 state
                    + (size_t)(8 * 16 * 40) * 2;                  // P scratch
  attn_scan<<<dim3(B_ * H_), 256, smem, stream>>>(qb, kbf, vbf, beta, att);

  // 4) output projection -> f32 d_out
  gemm_nt<false><<<ggrid, 256, 0, stream>>>(att, Wob, bo, (float*)d_out, M, E_, E_);
}
